// MultiScaleRetention_19215683682888
// MI455X (gfx1250) — matrix operand order repaired and checked by reading, not compiled, not run on hardware
//
#include <hip/hip_runtime.h>
#include <hip/hip_bf16.h>

// ---------------------------------------------------------------------------
// MultiScaleRetention forward for MI455X (gfx1250), wave32 + WMMA bf16
// + async global->LDS staged GEMM (ASYNCcnt pipeline).
// B=2 T=2048 E=1024 V=2048 H=8 Dk=128 Dv=256
// ---------------------------------------------------------------------------

#define BT     4096      // B*T
#define EDIM   1024
#define VDIM   2048
#define NHEAD  8
#define KDIM   128
#define HDIM   256
#define TLEN   2048

typedef __attribute__((ext_vector_type(16))) __bf16        v16bf;
typedef __attribute__((ext_vector_type(8)))  float         v8f;
typedef __attribute__((ext_vector_type(8)))  unsigned int  v8u;
typedef __attribute__((ext_vector_type(4)))  unsigned int  v4u;

__device__ __forceinline__ unsigned short f32_to_bf16(float f) {
    unsigned int u = __builtin_bit_cast(unsigned int, f);
    u += 0x7FFFu + ((u >> 16) & 1u);          // round-to-nearest-even
    return (unsigned short)(u >> 16);
}

// ------------------------------ converters ---------------------------------

__global__ void to_bf16_kernel(const float* __restrict__ src,
                               unsigned short* __restrict__ dst, int n) {
    int i = blockIdx.x * blockDim.x + threadIdx.x;
    if (i < n) dst[i] = f32_to_bf16(src[i]);
}

// src: fp32 [K][N] row-major  ->  dst: bf16 [N][K] row-major (transposed)
__global__ void transpose_bf16_kernel(const float* __restrict__ src,
                                      unsigned short* __restrict__ dst,
                                      int K, int N) {
    int i = blockIdx.x * blockDim.x + threadIdx.x;
    if (i >= K * N) return;
    int n = i / K;
    int k = i - n * K;
    dst[(size_t)n * K + k] = f32_to_bf16(src[(size_t)k * N + n]);
}

// ----------------------- theta shift (rotary-like) -------------------------
// in : fp32 [B*T][E]  (col = h*128 + d) -> out: bf16 [b][h][t][d]
__global__ void theta_shift_kernel(const float* __restrict__ q,
                                   const float* __restrict__ sin_t,
                                   const float* __restrict__ cos_t,
                                   unsigned short* __restrict__ dst,
                                   float scale) {
    int idx = blockIdx.x * blockDim.x + threadIdx.x;   // over BT*EDIM
    if (idx >= BT * EDIM) return;
    int col = idx & (EDIM - 1);
    int tg  = idx >> 10;          // global row
    int b   = tg >> 11;
    int t   = tg & (TLEN - 1);
    int h   = col >> 7;
    int d   = col & (KDIM - 1);
    float val = q[idx];
    float rot = (d & 1) ? q[idx - 1] : -q[idx + 1];
    float sn = sin_t[t * KDIM + d];
    float cs = cos_t[t * KDIM + d];
    float r  = scale * (val * cs + rot * sn);
    dst[(((size_t)(b * NHEAD + h) << 11) + t) * KDIM + d] = f32_to_bf16(r);
}

// v: fp32 [B*T][V] (col = h*256 + d)  ->  bf16 [b][h][d][t]
__global__ void v_transpose_kernel(const float* __restrict__ v,
                                   unsigned short* __restrict__ dst) {
    int idx = blockIdx.x * blockDim.x + threadIdx.x;   // over BT*VDIM
    if (idx >= BT * VDIM) return;
    int col = idx & (VDIM - 1);
    int tg  = idx >> 11;
    int b   = tg >> 11;
    int t   = tg & (TLEN - 1);
    int h   = col >> 8;
    int d   = col & (HDIM - 1);
    dst[(((size_t)((b * NHEAD + h) * HDIM + d)) << 11) + t] = f32_to_bf16(v[idx]);
}

// ------------------------------ WMMA GEMM ----------------------------------
// C[M][N] fp32 = A[M][K] bf16 @ Bt[N][K] bf16 (B pre-transposed).
// 128-thread (4-wave) workgroup computes a 64x64 C tile. A/B 64x32 tiles are
// staged global->LDS with global_load_async_to_lds_b128 (double-buffered,
// s_wait_asynccnt pipeline); each wave consumes a 16-row slice + all 64 cols.
// LDS rows padded to 80B: 16B-aligned b128 reads, conflict-free bank stride.

#define LDS_ROW_U16 40   // 32 bf16 payload + 8 pad (80 bytes)

__device__ __forceinline__ void async_tile_load(const unsigned short* gbase,
                                                int row_stride,
                                                unsigned lds_base, int tid) {
#pragma unroll
    for (int i = 0; i < 2; ++i) {
        int e   = (i << 7) + tid;       // 0..255 : 64 rows x 4 x 16B chunks
        int row = e >> 2;
        int c   = e & 3;
        unsigned long long src =
            (unsigned long long)(const void*)(gbase + (size_t)row * row_stride + c * 8);
        unsigned dst = lds_base + row * (LDS_ROW_U16 * 2) + c * 16;
        asm volatile("global_load_async_to_lds_b128 %0, %1, off"
                     :: "v"(dst), "v"(src) : "memory");
    }
}

__global__ void gemm_bf16_kernel(const unsigned short* __restrict__ A,
                                 const unsigned short* __restrict__ Bt,
                                 float* __restrict__ C, int M, int N, int K) {
    __shared__ __align__(16) unsigned short Abuf[2][64][LDS_ROW_U16];
    __shared__ __align__(16) unsigned short Bbuf[2][64][LDS_ROW_U16];

    const int tid  = threadIdx.x;
    const int w    = tid >> 5;
    const int lane = tid & 31;
    const int l16  = lane & 15;
    const int half = lane >> 4;
    const int mbase = blockIdx.y << 6;
    const int nbase = blockIdx.x << 6;

    const unsigned aoff[2] = { (unsigned)(uintptr_t)&Abuf[0][0][0],
                               (unsigned)(uintptr_t)&Abuf[1][0][0] };
    const unsigned boff[2] = { (unsigned)(uintptr_t)&Bbuf[0][0][0],
                               (unsigned)(uintptr_t)&Bbuf[1][0][0] };

    v8f acc[4] = {};

    // prologue: stage buffer 0 with the first K chunk
    async_tile_load(A  + (size_t)mbase * K, K, aoff[0], tid);
    async_tile_load(Bt + (size_t)nbase * K, K, boff[0], tid);

    int buf = 0;
    for (int kb = 0; kb < K; kb += 32) {
        if (kb + 32 < K) {
            async_tile_load(A  + (size_t)mbase * K + kb + 32, K, aoff[buf ^ 1], tid);
            async_tile_load(Bt + (size_t)nbase * K + kb + 32, K, boff[buf ^ 1], tid);
            // 4 newest async ops may stay in flight; older stage must be done
            asm volatile("s_wait_asynccnt 0x4" ::: "memory");
        } else {
            asm volatile("s_wait_asynccnt 0x0" ::: "memory");
        }
        __syncthreads();

        // A fragment: wave's 16-row slice, two contiguous b128 runs per lane
        const unsigned short* ar = &Abuf[buf][(w << 4) + l16][0];
        v4u alo = *(const v4u*)((const unsigned char*)ar + (half << 4));
        v4u ahi = *(const v4u*)((const unsigned char*)ar + 32 + (half << 4));
        v16bf a = __builtin_bit_cast(
            v16bf, __builtin_shufflevector(alo, ahi, 0, 1, 2, 3, 4, 5, 6, 7));
#pragma unroll
        for (int t = 0; t < 4; ++t) {
            const unsigned short* br = &Bbuf[buf][(t << 4) + l16][0];
            v4u blo = *(const v4u*)((const unsigned char*)br + (half << 4));
            v4u bhi = *(const v4u*)((const unsigned char*)br + 32 + (half << 4));
            v16bf bb = __builtin_bit_cast(
                v16bf, __builtin_shufflevector(blo, bhi, 0, 1, 2, 3, 4, 5, 6, 7));
            acc[t] = __builtin_amdgcn_wmma_f32_16x16x32_bf16(
                false, a, false, bb, (short)0, acc[t], false, false);
        }
        __syncthreads();
        buf ^= 1;
    }

    const int mrow = mbase + (w << 4);
#pragma unroll
    for (int t = 0; t < 4; ++t)
#pragma unroll
        for (int v = 0; v < 8; ++v)
            C[(size_t)(mrow + v + (half << 3)) * N + nbase + (t << 4) + l16] = acc[t][v];
}

// --------------------------- retention attention ---------------------------
// One wave per (b, h, 16-row tile): S = qr@kr^T, mask, deferred |S| denom,
// out += S_bf16 @ v, then denom clip + RMS norm + silu(g) -> bf16 act.
__global__ void retention_attn_kernel(const unsigned short* __restrict__ qr,
                                      const unsigned short* __restrict__ kr,
                                      const unsigned short* __restrict__ vbt,
                                      const float* __restrict__ mask,
                                      const float* __restrict__ g,
                                      unsigned short* __restrict__ att) {
    const int lane = threadIdx.x;
    const int l16  = lane & 15;
    const int half = lane >> 4;
    const int b = blockIdx.z, h = blockIdx.y, mt = blockIdx.x;
    const int mrow = mt << 4;

    const unsigned short* qbase = qr  + (((size_t)(b * NHEAD + h) << 11) + mrow) * KDIM;
    const unsigned short* kbase = kr  + (((size_t)(b * NHEAD + h) << 11)) * KDIM;
    const unsigned short* vbase = vbt + ((size_t)(b * NHEAD + h) << 19);   // 256*2048
    const float*          mbase = mask + ((size_t)h << 22);                // 2048*2048

    // A fragments of qr (16 x 128 bf16 = 4 chunks of 16x32), loaded once
    v16bf aq[4];
#pragma unroll
    for (int c = 0; c < 4; ++c) {
        v8u au;
#pragma unroll
        for (int v = 0; v < 8; ++v) {
            int k = (c << 5) + ((v >> 2) << 4) + (half << 3) + ((v & 3) << 1);
            au[v] = *(const unsigned int*)(qbase + (size_t)l16 * KDIM + k);
        }
        aq[c] = __builtin_bit_cast(v16bf, au);
    }

    v8f outacc[16] = {};
    float absacc[8] = {};

    __shared__ unsigned short slds[16 * 32];
    __shared__ float red[32 * 8];
    __shared__ float denomrow[16];
    __shared__ float scalerow[16];

    const int smax = mrow + 16;                 // causal: mask zeros s > t
    for (int sb = 0; sb < smax; sb += 32) {
        __builtin_prefetch(mbase + (size_t)(mrow + (half << 3)) * TLEN + sb, 0, 1);
        // ---- S = qr @ kr^T for s-tiles [sb, sb+16) and [sb+16, sb+32)
        v8f s0 = {}, s1 = {};
#pragma unroll
        for (int c = 0; c < 4; ++c) {
            v8u b0u, b1u;
#pragma unroll
            for (int v = 0; v < 8; ++v) {
                int k = (c << 5) + ((v >> 2) << 4) + (half << 3) + ((v & 3) << 1);
                b0u[v] = *(const unsigned int*)(kbase + (size_t)(sb + l16) * KDIM + k);
                b1u[v] = *(const unsigned int*)(kbase + (size_t)(sb + 16 + l16) * KDIM + k);
            }
            v16bf b0 = __builtin_bit_cast(v16bf, b0u);
            v16bf b1 = __builtin_bit_cast(v16bf, b1u);
            s0 = __builtin_amdgcn_wmma_f32_16x16x32_bf16(false, aq[c], false, b0, (short)0, s0, false, false);
            s1 = __builtin_amdgcn_wmma_f32_16x16x32_bf16(false, aq[c], false, b1, (short)0, s1, false, false);
        }
        // ---- mask multiply + |.| accumulate + spill to LDS as bf16 A-matrix
#pragma unroll
        for (int v = 0; v < 8; ++v) {
            int trow = mrow + v + (half << 3);
            float m0 = mbase[(size_t)trow * TLEN + sb + l16];
            float m1 = mbase[(size_t)trow * TLEN + sb + 16 + l16];
            float x0 = s0[v] * m0;
            float x1 = s1[v] * m1;
            absacc[v] += __builtin_fabsf(x0) + __builtin_fabsf(x1);
            slds[(v + (half << 3)) * 32 + l16]      = f32_to_bf16(x0);
            slds[(v + (half << 3)) * 32 + 16 + l16] = f32_to_bf16(x1);
        }
        __syncthreads();
        v8u su;
#pragma unroll
        for (int v = 0; v < 8; ++v) {
            int k = ((v >> 2) << 4) + (half << 3) + ((v & 3) << 1);
            su[v] = *(const unsigned int*)(slds + l16 * 32 + k);
        }
        v16bf sa = __builtin_bit_cast(v16bf, su);
        __syncthreads();
        // ---- out += S @ v   (16 n-tiles of Dv=256)
#pragma unroll
        for (int t = 0; t < 16; ++t) {
            v8u bu;
#pragma unroll
            for (int v = 0; v < 8; ++v) {
                int k = ((v >> 2) << 4) + (half << 3) + ((v & 3) << 1);
                bu[v] = *(const unsigned int*)(vbase + (size_t)((t << 4) + l16) * TLEN + sb + k);
            }
            v16bf bv = __builtin_bit_cast(v16bf, bu);
            outacc[t] = __builtin_amdgcn_wmma_f32_16x16x32_bf16(
                false, sa, false, bv, (short)0, outacc[t], false, false);
        }
    }

    // ---- denominator: per-row sum of |S*mask|, clipped to [1, 5e4]
#pragma unroll
    for (int v = 0; v < 8; ++v) red[lane * 8 + v] = absacc[v];
    __syncthreads();
    if (lane < 16) {
        int m = lane;
        float s = 0.f;
        for (int l = 0; l < 16; ++l) s += red[(((m >> 3) << 4) + l) * 8 + (m & 7)];
        denomrow[m] = fminf(fmaxf(s, 1.0f), 50000.0f);
    }
    __syncthreads();
    float sq[8] = {};
#pragma unroll
    for (int t = 0; t < 16; ++t)
#pragma unroll
        for (int v = 0; v < 8; ++v) {
            float o = outacc[t][v] / denomrow[v + (half << 3)];
            outacc[t][v] = o;
            sq[v] += o * o;
        }
    __syncthreads();
#pragma unroll
    for (int v = 0; v < 8; ++v) red[lane * 8 + v] = sq[v];
    __syncthreads();
    if (lane < 16) {
        int m = lane;
        float s = 0.f;
        for (int l = 0; l < 16; ++l) s += red[(((m >> 3) << 4) + l) * 8 + (m & 7)];
        scalerow[m] = rsqrtf(s * (1.0f / 256.0f) + 1e-6f);
    }
    __syncthreads();

    // ---- silu(g) gate + bf16 store to activation buffer [B*T][V]
    const float* gbase = g + ((size_t)((b << 11) + mrow)) * VDIM + h * HDIM;
    unsigned short* obase = att + ((size_t)((b << 11) + mrow)) * VDIM + h * HDIM;
#pragma unroll
    for (int t = 0; t < 16; ++t)
#pragma unroll
        for (int v = 0; v < 8; ++v) {
            int m = v + (half << 3);
            int d = (t << 4) + l16;
            float gv = gbase[(size_t)m * VDIM + d];
            float sg = gv / (1.0f + __expf(-gv));
            float o  = outacc[t][v] * scalerow[m] * sg;
            obase[(size_t)m * VDIM + d] = f32_to_bf16(o);
        }
}

// ------------------------------- launcher ----------------------------------

extern "C" void kernel_launch(void* const* d_in, const int* in_sizes, int n_in,
                              void* d_out, int out_size, void* d_ws, size_t ws_size,
                              hipStream_t stream) {
    const float* x    = (const float*)d_in[0];
    const float* sinp = (const float*)d_in[1];
    const float* cosp = (const float*)d_in[2];
    const float* mask = (const float*)d_in[3];
    const float* Wq   = (const float*)d_in[4];
    const float* Wk   = (const float*)d_in[5];
    const float* Wv   = (const float*)d_in[6];
    const float* Wg   = (const float*)d_in[7];
    const float* Wo   = (const float*)d_in[8];
    float* out = (float*)d_out;

    size_t off = 0;
    auto alloc = [&](size_t bytes) -> void* {
        void* p = (char*)d_ws + off;
        off += (bytes + 255) & ~(size_t)255;
        return p;
    };
    unsigned short* xb  = (unsigned short*)alloc((size_t)BT * EDIM * 2);
    unsigned short* Wqt = (unsigned short*)alloc((size_t)EDIM * EDIM * 2);
    unsigned short* Wkt = (unsigned short*)alloc((size_t)EDIM * EDIM * 2);
    unsigned short* Wvt = (unsigned short*)alloc((size_t)EDIM * VDIM * 2);
    unsigned short* Wgt = (unsigned short*)alloc((size_t)EDIM * VDIM * 2);
    unsigned short* Wot = (unsigned short*)alloc((size_t)VDIM * EDIM * 2);
    float*          prj = (float*)alloc((size_t)BT * VDIM * 4);   // reused q/k/v scratch
    float*          gbf = (float*)alloc((size_t)BT * VDIM * 4);
    unsigned short* qrb = (unsigned short*)alloc((size_t)BT * EDIM * 2);
    unsigned short* krb = (unsigned short*)alloc((size_t)BT * EDIM * 2);
    unsigned short* vbt = (unsigned short*)alloc((size_t)BT * VDIM * 2);
    unsigned short* att = (unsigned short*)alloc((size_t)BT * VDIM * 2);

    const float kscale = 0.08838834764831845f;   // KEY_DIM^-0.5

    to_bf16_kernel<<<(BT * EDIM) / 256, 256, 0, stream>>>(x, xb, BT * EDIM);
    transpose_bf16_kernel<<<(EDIM * EDIM) / 256, 256, 0, stream>>>(Wq, Wqt, EDIM, EDIM);
    transpose_bf16_kernel<<<(EDIM * EDIM) / 256, 256, 0, stream>>>(Wk, Wkt, EDIM, EDIM);
    transpose_bf16_kernel<<<(EDIM * VDIM) / 256, 256, 0, stream>>>(Wv, Wvt, EDIM, VDIM);
    transpose_bf16_kernel<<<(EDIM * VDIM) / 256, 256, 0, stream>>>(Wg, Wgt, EDIM, VDIM);
    transpose_bf16_kernel<<<(VDIM * EDIM) / 256, 256, 0, stream>>>(Wo, Wot, VDIM, EDIM);

    // q projection -> theta shift
    gemm_bf16_kernel<<<dim3(EDIM / 64, BT / 64), 128, 0, stream>>>(xb, Wqt, prj, BT, EDIM, EDIM);
    theta_shift_kernel<<<(BT * EDIM) / 256, 256, 0, stream>>>(prj, sinp, cosp, qrb, 1.0f);
    // k projection (scaling folded in) -> theta shift
    gemm_bf16_kernel<<<dim3(EDIM / 64, BT / 64), 128, 0, stream>>>(xb, Wkt, prj, BT, EDIM, EDIM);
    theta_shift_kernel<<<(BT * EDIM) / 256, 256, 0, stream>>>(prj, sinp, cosp, krb, kscale);
    // v projection -> [b,h,d,t] bf16
    gemm_bf16_kernel<<<dim3(VDIM / 64, BT / 64), 128, 0, stream>>>(xb, Wvt, prj, BT, VDIM, EDIM);
    v_transpose_kernel<<<(BT * VDIM) / 256, 256, 0, stream>>>(prj, vbt);
    // g projection (kept fp32, consumed in attention epilogue)
    gemm_bf16_kernel<<<dim3(VDIM / 64, BT / 64), 128, 0, stream>>>(xb, Wgt, gbf, BT, VDIM, EDIM);

    // retention core
    retention_attn_kernel<<<dim3(TLEN / 16, NHEAD, 2), 32, 0, stream>>>(
        qrb, krb, vbt, mask, gbf, att);

    // final projection -> fp32 output
    gemm_bf16_kernel<<<dim3(EDIM / 64, BT / 64), 128, 0, stream>>>(att, Wot, out, BT, EDIM, VDIM);
}
